// SelfAttention_42691974922421
// MI455X (gfx1250) — matrix operand order repaired and checked by reading, not compiled, not run on hardware
//
#include <hip/hip_runtime.h>
#include <hip/hip_bf16.h>
#include <stdint.h>

// Problem constants (from reference)
#define B_  2
#define T_  2048
#define D_  2048
#define H_  16
#define HD_ 128

typedef __bf16 bf16_t;
typedef __bf16 v16bf __attribute__((ext_vector_type(16)));
typedef float  v8f   __attribute__((ext_vector_type(8)));

union FragAB { v16bf v; uint4 q[2]; };   // 32 bytes: 16 bf16 per lane

// ---- gfx1250 async global->LDS copy (ASYNCcnt path), with safe fallback ---
#if defined(__has_builtin)
#  if __has_builtin(__builtin_amdgcn_global_load_async_to_lds_b128)
#    define HAVE_ASYNC_LDS 1
#  endif
#  if __has_builtin(__builtin_amdgcn_s_wait_asynccnt)
#    define HAVE_WAIT_ASYNC 1
#  endif
#endif

// Builtin expects pointers to <4 x i32>: global (AS1) src, LDS (AS3) dst.
typedef int v4i_t __attribute__((vector_size(16)));
typedef __attribute__((address_space(1))) v4i_t* gvec_p;
typedef __attribute__((address_space(3))) v4i_t* lvec_p;

__device__ __forceinline__ void cp16_async(const bf16_t* g, bf16_t* l) {
#if defined(HAVE_ASYNC_LDS)
    __builtin_amdgcn_global_load_async_to_lds_b128(
        (gvec_p)(void*)g, (lvec_p)(void*)l, 0, 0);
#else
    *reinterpret_cast<uint4*>(l) = *reinterpret_cast<const uint4*>(g);
#endif
}
__device__ __forceinline__ void wait_async() {
#if defined(HAVE_ASYNC_LDS)
#  if defined(HAVE_WAIT_ASYNC)
    __builtin_amdgcn_s_wait_asynccnt(0);
#  else
    asm volatile("s_wait_asynccnt 0" ::: "memory");
#  endif
#endif
}

__device__ __forceinline__ uint16_t f2bfbits(float f) {
    uint32_t u = __float_as_uint(f);
    u += 0x7FFFu + ((u >> 16) & 1u);          // round-to-nearest-even
    return (uint16_t)(u >> 16);
}
__device__ __forceinline__ bf16_t f2bf(float f) {
    uint16_t h = f2bfbits(f);
    bf16_t r; __builtin_memcpy(&r, &h, 2); return r;
}
__device__ __forceinline__ float bf2f(uint32_t bits16) {
    return __uint_as_float(bits16 << 16);
}

// ---------------------------------------------------------------------------
// Kernel 1a: fp32 -> bf16 conversion, 4 elements / thread (for x)
// ---------------------------------------------------------------------------
__global__ void f32_to_bf16_kernel(const float* __restrict__ in,
                                   bf16_t* __restrict__ out, int n4) {
    int i = blockIdx.x * blockDim.x + threadIdx.x;
    if (i >= n4) return;
    float4 f = reinterpret_cast<const float4*>(in)[i];
    uint2 p;
    p.x = (uint32_t)f2bfbits(f.x) | ((uint32_t)f2bfbits(f.y) << 16);
    p.y = (uint32_t)f2bfbits(f.z) | ((uint32_t)f2bfbits(f.w) << 16);
    reinterpret_cast<uint2*>(out)[i] = p;
}

// ---------------------------------------------------------------------------
// Kernel 1b: fp32 [K][N] -> bf16 [N][K] tiled transpose (for weights).
// Pre-transposing weights makes GEMM B-tile staging pure 16B copies.
// Block (32,8), 32x32 tile via LDS, coalesced both sides.
// ---------------------------------------------------------------------------
__global__ __launch_bounds__(256) void transpose_f32_to_bf16_kernel(
    const float* __restrict__ in, bf16_t* __restrict__ out, int K, int N)
{
    __shared__ float tile[32][33];
    const int k0 = blockIdx.y * 32;
    const int n0 = blockIdx.x * 32;
    const int tx = threadIdx.x;   // 0..31
    const int ty = threadIdx.y;   // 0..7
    #pragma unroll
    for (int i = 0; i < 4; ++i)
        tile[ty * 4 + i][tx] = in[(size_t)(k0 + ty * 4 + i) * N + n0 + tx];
    __syncthreads();
    #pragma unroll
    for (int i = 0; i < 4; ++i)
        out[(size_t)(n0 + ty * 4 + i) * K + k0 + tx] = f2bf(tile[tx][ty * 4 + i]);
}

// ---------------------------------------------------------------------------
// Kernel 2/5: bf16 GEMM  C[M,N] = A[M,K] * Bt[N,K]^T.
// 128x128x64 tiles, double-buffered LDS fed by async global->LDS copies,
// 8 waves (2x4), each wave 64x32 = 16 x v_wmma_f32_16x16x32_bf16 per K-step.
// ---------------------------------------------------------------------------
template <bool OUT_F32>
__global__ __launch_bounds__(256) void gemm_bf16_kernel(
    const bf16_t* __restrict__ A,   // [M,K] row-major
    const bf16_t* __restrict__ Bt,  // [N,K] row-major (pre-transposed B)
    void* __restrict__ C, int M, int N, int K)
{
    constexpr int BM = 128, BN = 128, BK = 64;
    constexpr int LD = BK + 8;      // 72 elems = 144B rows, 16B-aligned
    __shared__ bf16_t As[2][BM * LD];
    __shared__ bf16_t Bs[2][BN * LD];

    const int tid  = threadIdx.x;
    const int wave = tid >> 5;
    const int lane = tid & 31;
    const int l    = lane & 15;
    const int hi   = lane >> 4;         // 0: lanes 0-15, 1: lanes 16-31
    const int wm   = wave >> 2;         // 0..1 -> 64-row slab
    const int wn   = wave & 3;          // 0..3 -> 32-col slab
    const int m0   = blockIdx.y * BM;
    const int n0   = blockIdx.x * BN;

    // staging decomposition: thread -> (row 0..127, 32-col half)
    const int srow  = tid & 127;
    const int scol  = (tid >> 7) * 32;

    auto stage = [&](int kb, int buf) {
        const bf16_t* ga = A  + (size_t)(m0 + srow) * K + kb + scol;
        const bf16_t* gb = Bt + (size_t)(n0 + srow) * K + kb + scol;
        bf16_t* la = &As[buf][srow * LD + scol];
        bf16_t* lb = &Bs[buf][srow * LD + scol];
        #pragma unroll
        for (int j = 0; j < 4; ++j) {
            cp16_async(ga + j * 8, la + j * 8);
            cp16_async(gb + j * 8, lb + j * 8);
        }
    };

    const v8f z8 = {0.f,0.f,0.f,0.f,0.f,0.f,0.f,0.f};
    v8f acc[4][2];
    #pragma unroll
    for (int i = 0; i < 4; ++i) { acc[i][0] = z8; acc[i][1] = z8; }

    const int nIter = K / BK;
    stage(0, 0);
    wait_async();
    __syncthreads();

    for (int it = 0; it < nIter; ++it) {
        const int cur = it & 1;
        if (it + 1 < nIter) stage((it + 1) * BK, cur ^ 1);   // overlap with math

        #pragma unroll
        for (int s = 0; s < 2; ++s) {         // two K=32 slices per tile
            FragAB af[4], bfr[2];
            #pragma unroll
            for (int i = 0; i < 4; ++i) {     // A: lane -> row, chunks {hi*8, 16+hi*8}
                const bf16_t* p =
                    &As[cur][(wm * 64 + i * 16 + l) * LD + s * 32 + hi * 8];
                af[i].q[0] = *reinterpret_cast<const uint4*>(p);
                af[i].q[1] = *reinterpret_cast<const uint4*>(p + 16);
            }
            #pragma unroll
            for (int j = 0; j < 2; ++j) {     // B: lane -> col, chunks {hi*8, 16+hi*8}
                const uint4* p = reinterpret_cast<const uint4*>(
                    &Bs[cur][(wn * 32 + j * 16 + l) * LD + s * 32 + hi * 8]);
                bfr[j].q[0] = p[0]; bfr[j].q[1] = p[2];
            }
            #pragma unroll
            for (int i = 0; i < 4; ++i)
                #pragma unroll
                for (int j = 0; j < 2; ++j)
                    acc[i][j] = __builtin_amdgcn_wmma_f32_16x16x32_bf16(
                        false, af[i].v, false, bfr[j].v, (short)0, acc[i][j],
                        false, false);
        }

        if (it + 1 < nIter) {
            wait_async();       // next tile's async copies landed in LDS
            __syncthreads();    // publish; also guards ping-pong reuse
        }
    }

    // epilogue: C layout lane l -> col, VGPR r -> row r (+8 for hi half)
    #pragma unroll
    for (int i = 0; i < 4; ++i) {
        int row_base = m0 + wm * 64 + i * 16 + hi * 8;
        #pragma unroll
        for (int j = 0; j < 2; ++j) {
            int col = n0 + wn * 32 + j * 16 + l;
            #pragma unroll
            for (int r = 0; r < 8; ++r) {
                float vv = acc[i][j][r];
                if (OUT_F32)
                    ((float*)C)[(size_t)(row_base + r) * N + col] = vv;
                else
                    ((bf16_t*)C)[(size_t)(row_base + r) * N + col] = f2bf(vv);
            }
        }
    }
}

// ---------------------------------------------------------------------------
// Kernel 3: RoPE + head split.  qkv[B,T,3D] -> Q,K [B,H,T,HD] (roped),
// V transposed [B,H,HD,T] (so PV B-fragments are contiguous loads).
// ---------------------------------------------------------------------------
__global__ void rope_split_kernel(const bf16_t* __restrict__ qkv,
                                  bf16_t* __restrict__ Qr,
                                  bf16_t* __restrict__ Kr,
                                  bf16_t* __restrict__ Vt)
{
    const int NP = HD_ / 2;   // 64 pairs per head
    int idx = blockIdx.x * blockDim.x + threadIdx.x;
    int i = idx % NP;
    int t = (idx / NP) % T_;
    int h = (idx / (NP * T_)) % H_;
    int b =  idx / (NP * T_ * H_);

    size_t base = ((size_t)b * T_ + t) * (3 * D_);
    size_t off  = (size_t)h * HD_ + 2 * i;

    uint32_t qp = *reinterpret_cast<const uint32_t*>(qkv + base + off);
    uint32_t kp = *reinterpret_cast<const uint32_t*>(qkv + base + D_ + off);
    uint32_t vp = *reinterpret_cast<const uint32_t*>(qkv + base + 2 * D_ + off);

    float qe = bf2f(qp & 0xFFFFu), qo = bf2f(qp >> 16);
    float ke = bf2f(kp & 0xFFFFu), ko = bf2f(kp >> 16);

    float inv = __powf(10000.0f, -(float)(2 * i) / (float)HD_);
    float ang = (float)t * inv;
    float s, c; __sincosf(ang, &s, &c);

    uint32_t qrp = (uint32_t)f2bfbits(qe * c - qo * s)
                 | ((uint32_t)f2bfbits(qo * c + qe * s) << 16);
    uint32_t krp = (uint32_t)f2bfbits(ke * c - ko * s)
                 | ((uint32_t)f2bfbits(ko * c + ke * s) << 16);

    size_t qi = (((size_t)(b * H_ + h) * T_) + t) * HD_ + 2 * i;
    *reinterpret_cast<uint32_t*>(Qr + qi) = qrp;
    *reinterpret_cast<uint32_t*>(Kr + qi) = krp;

    size_t vt = ((size_t)(b * H_ + h) * HD_ + 2 * i) * T_ + t;
    bf16_t v0, v1;
    uint16_t v0b = (uint16_t)(vp & 0xFFFFu), v1b = (uint16_t)(vp >> 16);
    __builtin_memcpy(&v0, &v0b, 2); __builtin_memcpy(&v1, &v1b, 2);
    Vt[vt]      = v0;
    Vt[vt + T_] = v1;
}

// ---------------------------------------------------------------------------
// Kernel 4: causal flash attention (structure verified in round 1:
// clause'd b128 loads interleaved with wmma + partial loadcnt waits).
// ---------------------------------------------------------------------------
__global__ __launch_bounds__(256) void flash_attn_kernel(
    const bf16_t* __restrict__ Q,   // [B,H,T,HD]
    const bf16_t* __restrict__ Kk,  // [B,H,T,HD]
    const bf16_t* __restrict__ Vt,  // [B,H,HD,T]
    bf16_t* __restrict__ Y)         // [B,T,D]
{
    constexpr int BQ  = 128;
    constexpr int BKV = 64;
    constexpr int PLD = BKV + 8;    // 72 elems = 144B rows (16B-aligned)
    __shared__ bf16_t Plds[8][16 * PLD];   // per-wave private P staging

    const int tid  = threadIdx.x;
    const int wave = tid >> 5;
    const int lane = tid & 31;
    const int l    = lane & 15;
    const int hi   = lane >> 4;
    const int q0   = blockIdx.x * BQ;
    const int bh   = blockIdx.y;
    const int b    = bh / H_;
    const int h    = bh % H_;

    const bf16_t* Qh = Q  + (size_t)bh * T_ * HD_;
    const bf16_t* Kh = Kk + (size_t)bh * T_ * HD_;
    const bf16_t* Vh = Vt + (size_t)bh * HD_ * T_;

    const int qrow0 = q0 + wave * 16;

    FragAB qf[4];
    #pragma unroll
    for (int s = 0; s < 4; ++s) {
        const bf16_t* base = Qh + (size_t)(qrow0 + l) * HD_ + s * 32 + hi * 8;
        qf[s].q[0] = *reinterpret_cast<const uint4*>(base);
        qf[s].q[1] = *reinterpret_cast<const uint4*>(base + 16);
    }

    const v8f z8 = {0.f,0.f,0.f,0.f,0.f,0.f,0.f,0.f};
    v8f oacc[8];
    #pragma unroll
    for (int nt = 0; nt < 8; ++nt) oacc[nt] = z8;
    float mrun[8], lrun[8];
    #pragma unroll
    for (int r = 0; r < 8; ++r) { mrun[r] = -1e30f; lrun[r] = 0.f; }
    const float scale = 0.08838834764831845f;   // 1/sqrt(128)

    const int kv_end = q0 + BQ;                 // causal upper bound
    for (int kv0 = 0; kv0 < kv_end; kv0 += BKV) {
        v8f sacc[4];
        #pragma unroll
        for (int j = 0; j < 4; ++j) sacc[j] = z8;
        #pragma unroll
        for (int j = 0; j < 4; ++j) {
            #pragma unroll
            for (int s = 0; s < 4; ++s) {
                FragAB kf;
                const bf16_t* base =
                    Kh + (size_t)(kv0 + j * 16 + l) * HD_ + s * 32 + hi * 8;
                kf.q[0] = *reinterpret_cast<const uint4*>(base);
                kf.q[1] = *reinterpret_cast<const uint4*>(base + 16);
                sacc[j] = __builtin_amdgcn_wmma_f32_16x16x32_bf16(
                    false, qf[s].v, false, kf.v, (short)0, sacc[j],
                    false, false);
            }
        }

        float rmax[8];
        #pragma unroll
        for (int r = 0; r < 8; ++r) {
            int row = qrow0 + r + hi * 8;
            float mx = -1e30f;
            #pragma unroll
            for (int j = 0; j < 4; ++j) {
                int col = kv0 + j * 16 + l;
                float sv = sacc[j][r] * scale;
                sv = (col <= row) ? sv : -1e30f;
                sacc[j][r] = sv;
                mx = fmaxf(mx, sv);
            }
            #pragma unroll
            for (int off = 1; off < 16; off <<= 1)
                mx = fmaxf(mx, __shfl_xor(mx, off, 32));
            rmax[r] = mx;
        }
        float corr[8];
        #pragma unroll
        for (int r = 0; r < 8; ++r) {
            float mnew = fmaxf(mrun[r], rmax[r]);
            corr[r] = __expf(mrun[r] - mnew);
            mrun[r] = mnew;
        }

        bf16_t* Pw = &Plds[wave][0];
        float rsum[8];
        #pragma unroll
        for (int r = 0; r < 8; ++r) rsum[r] = 0.f;
        #pragma unroll
        for (int j = 0; j < 4; ++j) {
            #pragma unroll
            for (int r = 0; r < 8; ++r) {
                float p = __expf(sacc[j][r] - mrun[r]);
                rsum[r] += p;
                Pw[(r + hi * 8) * PLD + j * 16 + l] = f2bf(p);
            }
        }
        #pragma unroll
        for (int r = 0; r < 8; ++r) {
            float sum = rsum[r];
            #pragma unroll
            for (int off = 1; off < 16; off <<= 1)
                sum += __shfl_xor(sum, off, 32);
            lrun[r] = lrun[r] * corr[r] + sum;
        }

        #pragma unroll
        for (int nt = 0; nt < 8; ++nt)
            #pragma unroll
            for (int r = 0; r < 8; ++r)
                oacc[nt][r] *= corr[r];

        // reload P in A-layout (same-wave LDS ordering guarantees visibility)
        FragAB pf[2];
        #pragma unroll
        for (int s = 0; s < 2; ++s) {
            const bf16_t* base = Pw + (size_t)l * PLD + s * 32 + hi * 8;
            pf[s].q[0] = *reinterpret_cast<const uint4*>(base);
            pf[s].q[1] = *reinterpret_cast<const uint4*>(base + 16);
        }

        #pragma unroll
        for (int nt = 0; nt < 8; ++nt) {
            #pragma unroll
            for (int s = 0; s < 2; ++s) {
                FragAB vf;
                const bf16_t* base =
                    Vh + (size_t)(nt * 16 + l) * T_ + kv0 + s * 32 + hi * 8;
                vf.q[0] = *reinterpret_cast<const uint4*>(base);
                vf.q[1] = *reinterpret_cast<const uint4*>(base + 16);
                oacc[nt] = __builtin_amdgcn_wmma_f32_16x16x32_bf16(
                    false, pf[s].v, false, vf.v, (short)0, oacc[nt],
                    false, false);
            }
        }
    }

    #pragma unroll
    for (int nt = 0; nt < 8; ++nt) {
        int col = h * HD_ + nt * 16 + l;
        #pragma unroll
        for (int r = 0; r < 8; ++r) {
            int row = qrow0 + r + hi * 8;
            float v = oacc[nt][r] / lrun[r];
            Y[((size_t)b * T_ + row) * D_ + col] = f2bf(v);
        }
    }
}

// ---------------------------------------------------------------------------
// Host launcher
// ---------------------------------------------------------------------------
extern "C" void kernel_launch(void* const* d_in, const int* in_sizes, int n_in,
                              void* d_out, int out_size, void* d_ws, size_t ws_size,
                              hipStream_t stream) {
    const float* x      = (const float*)d_in[0];
    const float* w_qkv  = (const float*)d_in[1];
    const float* w_proj = (const float*)d_in[2];
    // d_in[3] = mask: causal, handled analytically in the flash kernel.
    float* out = (float*)d_out;

    char* ws = (char*)d_ws;
    auto alloc = [&](size_t bytes) -> char* {
        char* p = ws; ws += (bytes + 255) & ~(size_t)255; return p;
    };
    bf16_t* xb      = (bf16_t*)alloc((size_t)B_ * T_ * D_ * 2);
    bf16_t* wqkvT   = (bf16_t*)alloc((size_t)D_ * 3 * D_ * 2);  // [3D, D]
    bf16_t* wprojT  = (bf16_t*)alloc((size_t)D_ * D_ * 2);      // [D, D]
    bf16_t* qkv     = (bf16_t*)alloc((size_t)B_ * T_ * 3 * D_ * 2);
    bf16_t* Qr      = (bf16_t*)alloc((size_t)B_ * H_ * T_ * HD_ * 2);
    bf16_t* Kr      = (bf16_t*)alloc((size_t)B_ * H_ * T_ * HD_ * 2);
    bf16_t* Vt      = (bf16_t*)alloc((size_t)B_ * H_ * T_ * HD_ * 2);
    bf16_t* yb      = (bf16_t*)alloc((size_t)B_ * T_ * D_ * 2);

    // 1) x -> bf16; weights -> bf16 transposed [N][K]
    {
        int n4 = B_ * T_ * D_ / 4;
        f32_to_bf16_kernel<<<n4 / 256, 256, 0, stream>>>(x, xb, n4);
        dim3 blk(32, 8);
        dim3 g1(3 * D_ / 32, D_ / 32);
        transpose_f32_to_bf16_kernel<<<g1, blk, 0, stream>>>(w_qkv, wqkvT, D_, 3 * D_);
        dim3 g2(D_ / 32, D_ / 32);
        transpose_f32_to_bf16_kernel<<<g2, blk, 0, stream>>>(w_proj, wprojT, D_, D_);
    }
    // 2) QKV projection: [B*T, D] x [D, 3D] (B pre-transposed)
    {
        dim3 g(3 * D_ / 128, B_ * T_ / 128);
        gemm_bf16_kernel<false><<<g, 256, 0, stream>>>(
            xb, wqkvT, qkv, B_ * T_, 3 * D_, D_);
    }
    // 3) RoPE + split heads (+ V transpose)
    {
        int np = B_ * H_ * T_ * (HD_ / 2);
        rope_split_kernel<<<np / 256, 256, 0, stream>>>(qkv, Qr, Kr, Vt);
    }
    // 4) Causal flash attention
    {
        dim3 g(T_ / 128, B_ * H_);
        flash_attn_kernel<<<g, 256, 0, stream>>>(Qr, Kr, Vt, yb);
    }
    // 5) Output projection: [B*T, D] x [D, D] -> fp32 out
    {
        dim3 g(D_ / 128, B_ * T_ / 128);
        gemm_bf16_kernel<true><<<g, 256, 0, stream>>>(
            yb, wprojT, out, B_ * T_, D_, D_);
    }
}
